// HATBlock_37984690766441
// MI455X (gfx1250) — matrix operand order repaired and checked by reading, not compiled, not run on hardware
//
#include <hip/hip_runtime.h>
#include <hip/hip_bf16.h>
#include <stdint.h>

// ---------------------------------------------------------------------------
// Types / helpers
// ---------------------------------------------------------------------------
typedef __attribute__((ext_vector_type(16))) _Float16 v16h;
typedef __attribute__((ext_vector_type(8)))  float    v8f;
typedef __attribute__((ext_vector_type(4)))  unsigned int u32x4;
typedef __attribute__((ext_vector_type(8)))  int      i32x8;
typedef __attribute__((ext_vector_type(4)))  int      i32x4;

union FragH { v16h v; uint32_t u[8]; _Float16 h[16]; };
union U4 { u32x4 v; unsigned int e[4]; };
union I8 { i32x8 v; int e[8]; };

// Tensor Data Mover availability (probe-confirmed builtin names)
#if defined(__has_builtin)
# if __has_builtin(__builtin_amdgcn_tensor_load_to_lds) && \
     __has_builtin(__builtin_amdgcn_s_wait_tensorcnt)
#  define HAVE_TDM 1
# endif
#endif

__device__ __forceinline__ float gelu_tanh(float v) {
  // jax.nn.gelu (approximate=True)
  float inner = 0.7978845608028654f * (v + 0.044715f * v * v * v);
  return 0.5f * v * (1.0f + tanhf(inner));
}

// Model constants
#define DD   512
#define NH   16
#define HD   32
#define BB   16
#define NWIN 64
#define WS2  64
#define CT2  4
#define TOKW 68          // CT2 + WS2
#define TCT  256
#define HID_DIM 2048
// GEMM tiling
#define BM 128
#define BN 128
#define BK 32
// window-path chunking: 128 windows * 68 tokens = 8704 rows (divisible by BM)
#define CHUNK_WINS 128
#define CHUNK_TOK  (CHUNK_WINS * TOKW)
#define N_CHUNKS   8          // 1024 windows total

#ifdef HAVE_TDM
// Issue a TDM 2D tile load (global -> LDS), descriptor per CDNA5 ISA ch.8.
//   lds_off   : byte offset of destination tile within workgroup LDS
//   gaddr     : byte address of tile start within the tensor
//   tensor_d0 : tensor width  (elements, = row stride here)
//   tensor_d1 : tensor height (elements)
//   tile_d0   : tile width  (elements, contiguous)
//   tile_d1   : tile height (rows)
// data_size = 2 bytes (f16). Groups 2/3 zero (<=2D tensor).
__device__ __forceinline__ void tdm_load_tile_f16(uint32_t lds_off, const void* gaddr,
                                                  int tensor_d0, int tensor_d1,
                                                  int tile_d0, int tile_d1) {
  uint64_t ga = (uint64_t)(uintptr_t)gaddr;
  U4 g0;
  g0.e[0] = 1u;                                   // count=1 (valid descriptor)
  g0.e[1] = lds_off;                              // lds_addr (bytes)
  g0.e[2] = (unsigned int)ga;                     // global_addr[31:0]
  g0.e[3] = (unsigned int)((ga >> 32) & 0x1FFFFFFu) | (2u << 30);  // addr[56:32] | type=2
  I8 g1;
  g1.e[0] = (1 << 16);                            // workgroup_mask=0, data_size=1 (2B)
  g1.e[1] = (tensor_d0 & 0xFFFF) << 16;           // tensor_dim0[15:0]  (bits 63:48)
  g1.e[2] = ((tensor_d0 >> 16) & 0xFFFF) |        // tensor_dim0[31:16] (bits 79:64)
            ((tensor_d1 & 0xFFFF) << 16);         // tensor_dim1[15:0]  (bits 111:80)
  g1.e[3] = ((tensor_d1 >> 16) & 0xFFFF) |
            ((tile_d0 & 0xFFFF) << 16);           // tile_dim0 (bits 127:112)
  g1.e[4] = (tile_d1 & 0xFFFF);                   // tile_dim1; tile_dim2=0
  g1.e[5] = tensor_d0;                            // tensor_dim0_stride[31:0]
  g1.e[6] = 0;                                    // stride hi / dim1_stride lo
  g1.e[7] = 0;
  i32x4 z4 = (i32x4){0, 0, 0, 0};
#if (__clang_major__ >= 23)
  i32x8 z8 = (i32x8){0, 0, 0, 0, 0, 0, 0, 0};
  __builtin_amdgcn_tensor_load_to_lds(g0.v, g1.v, z4, z4, z8, 0);
#else
  __builtin_amdgcn_tensor_load_to_lds(g0.v, g1.v, z4, z4, 0);
#endif
}
#endif  // HAVE_TDM

// ---------------------------------------------------------------------------
// f32 -> f16 conversion
// ---------------------------------------------------------------------------
__global__ __launch_bounds__(256) void cvt_kernel(const float* __restrict__ src,
                                                  _Float16* __restrict__ dst, int n) {
  int i = blockIdx.x * 256 + threadIdx.x;
  if (i < n) dst[i] = (_Float16)src[i];
}

// ---------------------------------------------------------------------------
// Positional embedding MLP: out[p][d] = relu(coords @ w1 + b1) @ w2
// ---------------------------------------------------------------------------
__global__ __launch_bounds__(512) void posemb_kernel(const float* __restrict__ w1,
                                                     const float* __restrict__ b1,
                                                     const float* __restrict__ w2,
                                                     float* __restrict__ out, int side) {
  const int p = blockIdx.x;
  const int t = threadIdx.x;
  const int half = side / 2;
  const float cy = (float)(p / side - half) / (float)half;
  const float cx = (float)(p % side - half) / (float)half;
  __shared__ float hid[512];
  float hv = cy * w1[t] + cx * w1[512 + t] + b1[t];
  hid[t] = fmaxf(hv, 0.0f);
  __syncthreads();
  float acc = 0.0f;
#pragma unroll 8
  for (int j = 0; j < 512; ++j) acc += hid[j] * w2[(size_t)j * DD + t];
  out[(size_t)p * DD + t] = acc;
}

// ---------------------------------------------------------------------------
// Swin-v2 continuous position bias MLP table: h[(2ws-1)^2][16]
// ---------------------------------------------------------------------------
__global__ __launch_bounds__(512) void cpb_h_kernel(const float* __restrict__ w1,
                                                    const float* __restrict__ b1,
                                                    const float* __restrict__ w2,
                                                    float* __restrict__ hout, int ws) {
  const int n = blockIdx.x;
  const int t = threadIdx.x;
  const int L = 2 * ws - 1;
  float ry = (float)(n / L - (ws - 1));
  float rx = (float)(n % L - (ws - 1));
  float ty = ry / (float)(ws - 1) * 8.0f;
  float tx = rx / (float)(ws - 1) * 8.0f;
  float sy = (ty > 0.f) ? 1.f : ((ty < 0.f) ? -1.f : 0.f);
  float sx = (tx > 0.f) ? 1.f : ((tx < 0.f) ? -1.f : 0.f);
  const float inv_log2_8 = 1.0f / 3.0f;
  float fy = sy * log2f(fabsf(ty) + 1.0f) * inv_log2_8;
  float fx = sx * log2f(fabsf(tx) + 1.0f) * inv_log2_8;
  __shared__ float hid[512];
  float hv = fy * w1[t] + fx * w1[512 + t] + b1[t];
  hid[t] = fmaxf(hv, 0.0f);
  __syncthreads();
  if (t < NH) {
    float acc = 0.0f;
#pragma unroll 8
    for (int j = 0; j < 512; ++j) acc += hid[j] * w2[(size_t)j * NH + t];
    hout[(size_t)n * NH + t] = acc;
  }
}

// window bias, padded: out[16][68][68]
__global__ __launch_bounds__(256) void biasw_kernel(const float* __restrict__ h,
                                                    float* __restrict__ out) {
  int idx = blockIdx.x * 256 + threadIdx.x;
  const int total = NH * TOKW * TOKW;
  if (idx >= total) return;
  int head = idx / (TOKW * TOKW);
  int r = idx % (TOKW * TOKW);
  int i = r / TOKW, j = r % TOKW;
  float v = 0.0f;
  if (i >= CT2 && j >= CT2) {
    int q = i - CT2, k = j - CT2;
    int dy = (q / 8) - (k / 8) + 7;
    int dx = (q % 8) - (k % 8) + 7;
    float hv = h[(size_t)(dy * 15 + dx) * NH + head];
    v = 16.0f / (1.0f + __expf(-hv));
  }
  out[idx] = v;
}

// carrier bias: out[16][256][256]
__global__ __launch_bounds__(256) void biash_kernel(const float* __restrict__ h,
                                                    float* __restrict__ out) {
  int idx = blockIdx.x * 256 + threadIdx.x;
  int head = idx >> 16;
  int r = idx & 65535;
  int q = r >> 8, k = r & 255;
  int dy = (q / 16) - (k / 16) + 15;
  int dx = (q % 16) - (k % 16) + 15;
  float hv = h[(size_t)(dy * 31 + dx) * NH + head];
  out[idx] = 16.0f / (1.0f + __expf(-hv));
}

// ---------------------------------------------------------------------------
// Build carrier-token stream: ctg[b][t][d] = ct(permuted) + hpe[t][d]
// ---------------------------------------------------------------------------
__global__ __launch_bounds__(256) void buildctg_kernel(const float* __restrict__ ct,
                                                       const float* __restrict__ hpe,
                                                       float* __restrict__ ctg) {
  size_t idx = (size_t)blockIdx.x * 256 + threadIdx.x;  // B*256*512
  int d = idx & 511;
  size_t r = idx >> 9;
  int t = (int)(r & 255);
  int b = (int)(r >> 8);
  int cx = t & 1, sx = (t >> 1) & 7, cy = (t >> 4) & 1, sy = (t >> 5) & 7;
  int w = sy * 8 + sx, jj = cy * 2 + cx;
  ctg[idx] = ct[(((size_t)b * NWIN + w) * CT2 + jj) * DD + d] + hpe[(size_t)t * DD + d];
}

// ---------------------------------------------------------------------------
// Concat: X[b][w][j][d] = j<4 ? ctg : x + pe     (fully overwrites d_out)
// ---------------------------------------------------------------------------
__global__ __launch_bounds__(256) void concat_kernel(const float* __restrict__ x,
                                                     const float* __restrict__ pe,
                                                     const float* __restrict__ ctg,
                                                     float* __restrict__ X) {
  size_t idx = (size_t)blockIdx.x * 256 + threadIdx.x;  // B*64*68*512
  int d = idx & 511;
  size_t r = idx >> 9;
  int j = (int)(r % TOKW);
  size_t r2 = r / TOKW;
  int w = (int)(r2 & 63);
  int b = (int)(r2 >> 6);
  float v;
  if (j < CT2) {
    int cy = j >> 1, cx = j & 1, sy = w >> 3, sx = w & 7;
    int t = sy * 32 + cy * 16 + sx * 2 + cx;
    v = ctg[((size_t)b * TCT + t) * DD + d];
  } else {
    int p = j - CT2;
    v = x[(((size_t)b * NWIN + w) * WS2 + p) * DD + d] + pe[(size_t)p * DD + d];
  }
  X[idx] = v;
}

// ---------------------------------------------------------------------------
// LayerNorm over D=512, one token per 256-thread block, f16 output
// ---------------------------------------------------------------------------
__global__ __launch_bounds__(256) void ln_kernel(const float* __restrict__ x,
                                                 const float* __restrict__ sc,
                                                 const float* __restrict__ bi,
                                                 _Float16* __restrict__ out) {
  const int tok = blockIdx.x;
  const int t = threadIdx.x;
  const float* xr = x + (size_t)tok * DD;
  float a0 = xr[t], a1 = xr[t + 256];
  __shared__ float red[256];
  red[t] = a0 + a1;
  __syncthreads();
  for (int s = 128; s > 0; s >>= 1) { if (t < s) red[t] += red[t + s]; __syncthreads(); }
  float mean = red[0] * (1.0f / 512.0f);
  __syncthreads();
  float d0 = a0 - mean, d1 = a1 - mean;
  red[t] = d0 * d0 + d1 * d1;
  __syncthreads();
  for (int s = 128; s > 0; s >>= 1) { if (t < s) red[t] += red[t + s]; __syncthreads(); }
  float rstd = rsqrtf(red[0] * (1.0f / 512.0f) + 1e-5f);
  _Float16* orow = out + (size_t)tok * DD;
  orow[t]       = (_Float16)(d0 * rstd * sc[t] + bi[t]);
  orow[t + 256] = (_Float16)(d1 * rstd * sc[t + 256] + bi[t + 256]);
}

// ---------------------------------------------------------------------------
// WMMA GEMM: out[M,N] = A[M,K](f16) * W[K,N](f16) + bias (+gelu) (+residual)
// 256 threads = 8 wave32; block tile 128x128; each wave: 16-row strip x 128 col
// A tile arrives via the Tensor Data Mover (TDM) when available; B tile is
// loaded + transposed into LDS manually (TDM cannot transpose).
// A/B fragments use contiguous k order, k = 16*(lane/16) + s. Requires M%128==0, N%128==0, K%32==0.
// ---------------------------------------------------------------------------
#define ASTRIDE BK            // 32 halves, unpadded (TDM writes dense rows)
__global__ __launch_bounds__(256) void gemm_kernel(
    const _Float16* __restrict__ A, const _Float16* __restrict__ W,
    const float* __restrict__ bias, const float* __restrict__ resid,
    float* __restrict__ outF, _Float16* __restrict__ outH,
    int M, int N, int K, int act) {
  __shared__ __align__(16) _Float16 As[BM][ASTRIDE];  // row-major, stride 32
  __shared__ __align__(16) _Float16 Bs[BN][BK + 4];   // transposed [col][k], stride 36
  const int t = threadIdx.x;
  const int wave = t >> 5, lane = t & 31;
  const int row0 = blockIdx.x * BM;
  const int col0 = blockIdx.y * BN;

  v8f acc[8];
#pragma unroll
  for (int g = 0; g < 8; ++g) acc[g] = (v8f){0.f,0.f,0.f,0.f,0.f,0.f,0.f,0.f};

  for (int k0 = 0; k0 < K; k0 += BK) {
    __syncthreads();
    // ---- A tile (128x32 halves)
#ifdef HAVE_TDM
    if (wave == 0) {
      uint32_t lds_off =
          (uint32_t)(uintptr_t)(__attribute__((address_space(3))) char*)&As[0][0];
      tdm_load_tile_f16(lds_off, A + (size_t)row0 * K + k0,
                        /*tensor_d0=*/K, /*tensor_d1=*/M,
                        /*tile_d0=*/BK, /*tile_d1=*/BM);
      __builtin_amdgcn_s_wait_tensorcnt(0);
    }
#else
#pragma unroll
    for (int s = 0; s < 2; ++s) {
      int seg = t + s * 256;
      int r = seg >> 2;
      int off = (seg & 3) * 8;
      const float4 d4 = *(const float4*)(A + (size_t)(row0 + r) * K + k0 + off);
      *(float4*)&As[r][off] = d4;
    }
#endif
    // ---- B tile (32x128): load row-major, store transposed into LDS
#pragma unroll
    for (int s = 0; s < 2; ++s) {
      int seg = t + s * 256;
      int r = seg >> 4;            // k row 0..31
      int off = (seg & 15) * 8;    // col 0..120
      const float4 d4 = *(const float4*)(W + (size_t)(k0 + r) * N + col0 + off);
      const _Float16* hh = (const _Float16*)&d4;
#pragma unroll
      for (int e = 0; e < 8; ++e) Bs[off + e][r] = hh[e];
    }
    // prefetch next K tile
    if (k0 + BK < K) {
      __builtin_prefetch(A + (size_t)(row0 + (t >> 1)) * K + k0 + BK, 0, 0);
      __builtin_prefetch(W + (size_t)(k0 + BK + (t >> 3)) * N + col0, 0, 0);
    }
    __syncthreads();

    // ---- A fragment for this wave's 16-row strip (16-bit A 16x32 layout)
    FragH af;
    {
      int m = (wave << 4) + (lane & 15);
      int kb = (lane >> 4) << 4;   // 0 or 16
#pragma unroll
      for (int v = 0; v < 8; ++v) {
        int k = kb + 2 * v;
        af.u[v] = *(const uint32_t*)&As[m][k];
      }
    }
    // ---- 8 column groups (16-bit B 32x16 layout)
#pragma unroll
    for (int g = 0; g < 8; ++g) {
      FragH bf;
      int n = (g << 4) + (lane & 15);
      int kb = (lane >> 4) << 4;   // 0 or 16
#pragma unroll
      for (int v = 0; v < 8; ++v)
        bf.u[v] = *(const uint32_t*)&Bs[n][kb + 2 * v];
      acc[g] = __builtin_amdgcn_wmma_f32_16x16x32_f16(
          false, af.v, false, bf.v, (short)0, acc[g], false, false);
    }
  }

  // ---- epilogue: C layout: VGPR r -> M = 16*wave + r + 8*(lane>>4), N = lane&15
  const int mbase = row0 + (wave << 4) + ((lane >> 4) << 3);
  const int nloc = lane & 15;
#pragma unroll
  for (int g = 0; g < 8; ++g) {
    int col = col0 + (g << 4) + nloc;
    float bv = bias ? bias[col] : 0.0f;
#pragma unroll
    for (int r = 0; r < 8; ++r) {
      size_t o = (size_t)(mbase + r) * N + col;
      float v = acc[g][r] + bv;
      if (act == 1) v = gelu_tanh(v);
      if (resid) v += resid[o];
      if (outF) outF[o] = v;
      if (outH) outH[o] = (_Float16)v;
    }
  }
}

// ---------------------------------------------------------------------------
// Carrier attention: grid (B, H), 256 threads = 256 queries, S=256, hd=32
// ---------------------------------------------------------------------------
__global__ __launch_bounds__(256) void attn_c_kernel(const _Float16* __restrict__ qkv,
                                                     const float* __restrict__ bias,
                                                     _Float16* __restrict__ out) {
  const int b = blockIdx.x, h = blockIdx.y;
  const int t = threadIdx.x;
  __shared__ _Float16 Ks[TCT][HD + 1];
  __shared__ _Float16 Vs[TCT][HD + 1];
  const size_t base = ((size_t)b * TCT + t) * (3 * DD);
#pragma unroll 8
  for (int i = 0; i < HD; ++i) {
    Ks[t][i] = qkv[base + DD + h * HD + i];
    Vs[t][i] = qkv[base + 2 * DD + h * HD + i];
  }
  __syncthreads();
  float q[HD];
  const float scale = 0.17677669529663687f;  // 1/sqrt(32)
#pragma unroll 8
  for (int i = 0; i < HD; ++i) q[i] = (float)qkv[base + h * HD + i] * scale;
  float m = -1e30f, l = 0.0f, o[HD];
#pragma unroll 8
  for (int i = 0; i < HD; ++i) o[i] = 0.0f;
  const float* brow = bias + ((size_t)h * TCT + t) * TCT;
  for (int kk = 0; kk < TCT; ++kk) {
    float s = brow[kk];
#pragma unroll 8
    for (int i = 0; i < HD; ++i) s += q[i] * (float)Ks[kk][i];
    float nm = fmaxf(m, s);
    float corr = __expf(m - nm);
    float p = __expf(s - nm);
    l = l * corr + p;
#pragma unroll 8
    for (int i = 0; i < HD; ++i) o[i] = o[i] * corr + p * (float)Vs[kk][i];
    m = nm;
  }
  float inv = 1.0f / l;
  _Float16* op = out + ((size_t)b * TCT + t) * DD + h * HD;
#pragma unroll 8
  for (int i = 0; i < HD; ++i) op[i] = (_Float16)(o[i] * inv);
}

// ---------------------------------------------------------------------------
// Window attention: grid (windows_in_chunk, H), 96 threads (t<68 = queries)
// ---------------------------------------------------------------------------
__global__ __launch_bounds__(96) void attn_w_kernel(const _Float16* __restrict__ qkv,
                                                    const float* __restrict__ bias,
                                                    _Float16* __restrict__ out) {
  const int w = blockIdx.x, h = blockIdx.y;
  const int t = threadIdx.x;
  __shared__ _Float16 Ks[TOKW][HD + 1];
  __shared__ _Float16 Vs[TOKW][HD + 1];
  size_t base = 0;
  if (t < TOKW) {
    base = ((size_t)w * TOKW + t) * (3 * DD);
#pragma unroll 8
    for (int i = 0; i < HD; ++i) {
      Ks[t][i] = qkv[base + DD + h * HD + i];
      Vs[t][i] = qkv[base + 2 * DD + h * HD + i];
    }
  }
  __syncthreads();
  if (t >= TOKW) return;
  float q[HD];
  const float scale = 0.17677669529663687f;
#pragma unroll 8
  for (int i = 0; i < HD; ++i) q[i] = (float)qkv[base + h * HD + i] * scale;
  float m = -1e30f, l = 0.0f, o[HD];
#pragma unroll 8
  for (int i = 0; i < HD; ++i) o[i] = 0.0f;
  const float* brow = bias + ((size_t)h * TOKW + t) * TOKW;
  for (int kk = 0; kk < TOKW; ++kk) {
    float s = brow[kk];
#pragma unroll 8
    for (int i = 0; i < HD; ++i) s += q[i] * (float)Ks[kk][i];
    float nm = fmaxf(m, s);
    float corr = __expf(m - nm);
    float p = __expf(s - nm);
    l = l * corr + p;
#pragma unroll 8
    for (int i = 0; i < HD; ++i) o[i] = o[i] * corr + p * (float)Vs[kk][i];
    m = nm;
  }
  float inv = 1.0f / l;
  _Float16* op = out + ((size_t)w * TOKW + t) * DD + h * HD;
#pragma unroll 8
  for (int i = 0; i < HD; ++i) op[i] = (_Float16)(o[i] * inv);
}

// ---------------------------------------------------------------------------
// Host orchestration
// ---------------------------------------------------------------------------
extern "C" void kernel_launch(void* const* d_in, const int* in_sizes, int n_in,
                              void* d_out, int out_size, void* d_ws, size_t ws_size,
                              hipStream_t stream) {
  const float* x      = (const float*)d_in[0];
  const float* ct     = (const float*)d_in[1];
  const float* pe_w1  = (const float*)d_in[2];
  const float* pe_b1  = (const float*)d_in[3];
  const float* pe_w2  = (const float*)d_in[4];
  const float* hpe_w1 = (const float*)d_in[5];
  const float* hpe_b1 = (const float*)d_in[6];
  const float* hpe_w2 = (const float*)d_in[7];
  const float* n1_s   = (const float*)d_in[8];
  const float* n1_b   = (const float*)d_in[9];
  const float* n2_s   = (const float*)d_in[10];
  const float* n2_b   = (const float*)d_in[11];
  const float* hn1_s  = (const float*)d_in[12];
  const float* hn1_b  = (const float*)d_in[13];
  const float* hn2_s  = (const float*)d_in[14];
  const float* hn2_b  = (const float*)d_in[15];
  const float* qkv_w  = (const float*)d_in[16];
  const float* qkv_b  = (const float*)d_in[17];
  const float* proj_w = (const float*)d_in[18];
  const float* proj_b = (const float*)d_in[19];
  const float* cpb_w1 = (const float*)d_in[20];
  const float* cpb_b1 = (const float*)d_in[21];
  const float* cpb_w2 = (const float*)d_in[22];
  const float* hqkv_w = (const float*)d_in[23];
  const float* hqkv_b = (const float*)d_in[24];
  const float* hproj_w= (const float*)d_in[25];
  const float* hproj_b= (const float*)d_in[26];
  const float* hcpb_w1= (const float*)d_in[27];
  const float* hcpb_b1= (const float*)d_in[28];
  const float* hcpb_w2= (const float*)d_in[29];
  const float* fc1_w  = (const float*)d_in[30];
  const float* fc1_b  = (const float*)d_in[31];
  const float* fc2_w  = (const float*)d_in[32];
  const float* fc2_b  = (const float*)d_in[33];
  const float* hfc1_w = (const float*)d_in[34];
  const float* hfc1_b = (const float*)d_in[35];
  const float* hfc2_w = (const float*)d_in[36];
  const float* hfc2_b = (const float*)d_in[37];

  // workspace bump allocator (256B aligned)
  char* wsp = (char*)d_ws;
  size_t off = 0;
  auto alloc = [&](size_t bytes) -> void* {
    void* p = wsp + off;
    off = (off + bytes + 255) & ~(size_t)255;
    return p;
  };

  _Float16* w16_qkv  = (_Float16*)alloc((size_t)DD * 3 * DD * 2);
  _Float16* w16_proj = (_Float16*)alloc((size_t)DD * DD * 2);
  _Float16* w16_fc1  = (_Float16*)alloc((size_t)DD * HID_DIM * 2);
  _Float16* w16_fc2  = (_Float16*)alloc((size_t)HID_DIM * DD * 2);
  _Float16* w16_hqkv = (_Float16*)alloc((size_t)DD * 3 * DD * 2);
  _Float16* w16_hproj= (_Float16*)alloc((size_t)DD * DD * 2);
  _Float16* w16_hfc1 = (_Float16*)alloc((size_t)DD * HID_DIM * 2);
  _Float16* w16_hfc2 = (_Float16*)alloc((size_t)HID_DIM * DD * 2);
  float* pe_tab  = (float*)alloc((size_t)WS2 * DD * 4);
  float* hpe_tab = (float*)alloc((size_t)TCT * DD * 4);
  float* h_small = (float*)alloc((size_t)15 * 15 * NH * 4);
  float* h_big   = (float*)alloc((size_t)31 * 31 * NH * 4);
  float* bias_w  = (float*)alloc((size_t)NH * TOKW * TOKW * 4);
  float* bias_h  = (float*)alloc((size_t)NH * TCT * TCT * 4);
  float* ctg     = (float*)alloc((size_t)BB * TCT * DD * 4);
  // big chunk buffers (also reused by the smaller carrier path)
  _Float16* lnh  = (_Float16*)alloc((size_t)CHUNK_TOK * DD * 2);
  _Float16* qkvh = (_Float16*)alloc((size_t)CHUNK_TOK * 3 * DD * 2);
  _Float16* attnh= (_Float16*)alloc((size_t)CHUNK_TOK * DD * 2);
  _Float16* hidh = (_Float16*)alloc((size_t)CHUNK_TOK * HID_DIM * 2);
  (void)ws_size; (void)in_sizes; (void)n_in; (void)out_size;

  auto cvt = [&](const float* s, _Float16* d, int n) {
    cvt_kernel<<<(n + 255) / 256, 256, 0, stream>>>(s, d, n);
  };
  cvt(qkv_w,  w16_qkv,  DD * 3 * DD);
  cvt(proj_w, w16_proj, DD * DD);
  cvt(fc1_w,  w16_fc1,  DD * HID_DIM);
  cvt(fc2_w,  w16_fc2,  HID_DIM * DD);
  cvt(hqkv_w, w16_hqkv, DD * 3 * DD);
  cvt(hproj_w,w16_hproj,DD * DD);
  cvt(hfc1_w, w16_hfc1, DD * HID_DIM);
  cvt(hfc2_w, w16_hfc2, HID_DIM * DD);

  // pos-emb + bias tables
  posemb_kernel<<<WS2, 512, 0, stream>>>(pe_w1, pe_b1, pe_w2, pe_tab, 8);
  posemb_kernel<<<TCT, 512, 0, stream>>>(hpe_w1, hpe_b1, hpe_w2, hpe_tab, 16);
  cpb_h_kernel<<<15 * 15, 512, 0, stream>>>(cpb_w1, cpb_b1, cpb_w2, h_small, 8);
  cpb_h_kernel<<<31 * 31, 512, 0, stream>>>(hcpb_w1, hcpb_b1, hcpb_w2, h_big, 16);
  biasw_kernel<<<(NH * TOKW * TOKW + 255) / 256, 256, 0, stream>>>(h_small, bias_w);
  biash_kernel<<<(NH * TCT * TCT) / 256, 256, 0, stream>>>(h_big, bias_h);

  auto gemm = [&](const _Float16* A, const _Float16* W, const float* bias,
                  const float* resid, float* oF, _Float16* oH,
                  int M, int N, int K, int act) {
    dim3 grid(M / BM, N / BN);
    gemm_kernel<<<grid, 256, 0, stream>>>(A, W, bias, resid, oF, oH, M, N, K, act);
  };

  // ---------------- carrier-token path (4096 tokens) ----------------
  const int MC = BB * TCT;  // 4096
  buildctg_kernel<<<(MC * DD) / 256, 256, 0, stream>>>(ct, hpe_tab, ctg);
  ln_kernel<<<MC, 256, 0, stream>>>(ctg, hn1_s, hn1_b, lnh);
  gemm(lnh, w16_hqkv, hqkv_b, nullptr, nullptr, qkvh, MC, 3 * DD, DD, 0);
  { dim3 g(BB, NH); attn_c_kernel<<<g, 256, 0, stream>>>(qkvh, bias_h, attnh); }
  gemm(attnh, w16_hproj, hproj_b, ctg, ctg, nullptr, MC, DD, DD, 0);
  ln_kernel<<<MC, 256, 0, stream>>>(ctg, hn2_s, hn2_b, lnh);
  gemm(lnh, w16_hfc1, hfc1_b, nullptr, nullptr, hidh, MC, HID_DIM, DD, 1);
  gemm(hidh, w16_hfc2, hfc2_b, ctg, ctg, nullptr, MC, DD, HID_DIM, 0);

  // ---------------- concat carrier tokens + x + pe into d_out ----------------
  float* X = (float*)d_out;  // residual stream: (B*NW*68, 512)
  concat_kernel<<<(size_t)(BB * NWIN * TOKW * DD) / 256, 256, 0, stream>>>(
      x, pe_tab, ctg, X);

  // ---------------- window path, 8 chunks of 128 windows ----------------
  for (int c = 0; c < N_CHUNKS; ++c) {
    float* Xc = X + (size_t)c * CHUNK_TOK * DD;
    // attention sub-block
    ln_kernel<<<CHUNK_TOK, 256, 0, stream>>>(Xc, n1_s, n1_b, lnh);
    gemm(lnh, w16_qkv, qkv_b, nullptr, nullptr, qkvh, CHUNK_TOK, 3 * DD, DD, 0);
    { dim3 g(CHUNK_WINS, NH); attn_w_kernel<<<g, 96, 0, stream>>>(qkvh, bias_w, attnh); }
    gemm(attnh, w16_proj, proj_b, Xc, Xc, nullptr, CHUNK_TOK, DD, DD, 0);
    // MLP sub-block
    ln_kernel<<<CHUNK_TOK, 256, 0, stream>>>(Xc, n2_s, n2_b, lnh);
    gemm(lnh, w16_fc1, fc1_b, nullptr, nullptr, hidh, CHUNK_TOK, HID_DIM, DD, 1);
    gemm(hidh, w16_fc2, fc2_b, Xc, Xc, nullptr, CHUNK_TOK, DD, HID_DIM, 0);
  }
}
